// FeedForwardQuantum_65481071395761
// MI455X (gfx1250) — matrix operand order repaired and checked by reading, not compiled, not run on hardware
//
#include <hip/hip_runtime.h>

typedef __attribute__((ext_vector_type(16))) _Float16 v16h;
typedef __attribute__((ext_vector_type(8)))  _Float16 v8h;
typedef __attribute__((ext_vector_type(4)))  _Float16 v4h;
typedef __attribute__((ext_vector_type(2)))  _Float16 v2h;
typedef __attribute__((ext_vector_type(8)))  float    v8f;

#define BB    8
#define SS    2048
#define EE    512
#define FFN   2048
#define QQ    8
#define MTOT  (BB * SS)      // 16384 rows
#define MTILE 128            // rows per workgroup (4 M-subtiles per wave)
#define KCH   64             // FFN chunk per iteration
#define HPAD  (KCH + 8)      // LDS row stride in halves (72 -> 144B, 16B aligned, bank-spread)

// ---------------------------------------------------------------------------
// Kernel 0: one-time convert w2 [E][FFN] f32 -> f16 in workspace (2 MB).
// Halves L2 read traffic for the big GEMM and matches WMMA input type.
// ---------------------------------------------------------------------------
__global__ __launch_bounds__(256) void cvt_w2_kernel(const float* __restrict__ w2,
                                                     _Float16* __restrict__ w2h) {
  int i = (blockIdx.x * 256 + threadIdx.x) * 4;   // E*FFN is an exact multiple of 1024
  float4 v = *(const float4*)(w2 + i);
  v4h p;
  p[0] = (_Float16)v.x; p[1] = (_Float16)v.y;
  p[2] = (_Float16)v.z; p[3] = (_Float16)v.w;
  *(v4h*)(w2h + i) = p;
}

// ---------------------------------------------------------------------------
// Kernel 1: fused  quantum-layer -> fc1+relu (LDS f16) -> fc2 (WMMA f16).
// One workgroup (8 wave32) per 128 rows; full E=512 output width.
// Wave blocking 4M x 8N: each B fragment feeds 4 WMMAs from registers,
// each A fragment feeds 8; each w2 element read only 2x per workgroup.
// ---------------------------------------------------------------------------
__global__ __launch_bounds__(256) void ffnq_kernel(
    const float* __restrict__ x,  const float* __restrict__ ry,
    const float* __restrict__ w1, const float* __restrict__ b1,
    const _Float16* __restrict__ w2h, const float* __restrict__ b2,
    float* __restrict__ out) {

  __shared__ float    zT[MTILE][QQ];       // cos(x)*cos(ry), f32            (4 KB)
  __shared__ _Float16 hT[MTILE][HPAD];     // relu(fc1) chunk, f16, padded  (18 KB)

  const int t  = threadIdx.x;
  const int m0 = blockIdx.x * MTILE;

  // ---- Phase 0: quantum layer (closed form <Z_i> = cos(x_i)*cos(ry_i)) ----
  if (t < MTILE) {
    const float* xr = x + (size_t)(m0 + t) * EE;   // only first 8 of 512 cols needed
    float4 x0 = *(const float4*)(xr);
    float4 x1 = *(const float4*)(xr + 4);
    zT[t][0] = __cosf(x0.x) * __cosf(ry[0]);
    zT[t][1] = __cosf(x0.y) * __cosf(ry[1]);
    zT[t][2] = __cosf(x0.z) * __cosf(ry[2]);
    zT[t][3] = __cosf(x0.w) * __cosf(ry[3]);
    zT[t][4] = __cosf(x1.x) * __cosf(ry[4]);
    zT[t][5] = __cosf(x1.y) * __cosf(ry[5]);
    zT[t][6] = __cosf(x1.z) * __cosf(ry[6]);
    zT[t][7] = __cosf(x1.w) * __cosf(ry[7]);
  }
  __syncthreads();

  const int wave   = t >> 5;             // 0..7 (wave32)
  const int lane   = t & 31;
  const int lane16 = lane & 15;
  const int laneHi = lane >> 4;          // 0 = lanes 0-15, 1 = lanes 16-31
  const int mBase  = (wave >> 2) * 64;   // four 16-row subtiles: mBase + {0,16,32,48}
  const int eQ     = (wave & 3) * 128;   // which 128-col quarter of E (8 N-subtiles)

  // fc1 work decomposition: thread -> (row m, f-half g of 32 cols)
  const int m = t & (MTILE - 1);
  const int g = t >> 7;                  // 0..1

  v8f zero;
#pragma unroll
  for (int k = 0; k < 8; ++k) zero[k] = 0.0f;
  v8f acc[4][8];
#pragma unroll
  for (int mi = 0; mi < 4; ++mi)
#pragma unroll
    for (int j = 0; j < 8; ++j) acc[mi][j] = zero;

  for (int fc = 0; fc < FFN; fc += KCH) {
    // ---- Phase 1: hT[128][64] = relu(z @ w1^T + b1), f16 into LDS ----
#pragma unroll
    for (int p = 0; p < 16; ++p) {
      const int   fl  = g * 32 + p * 2;
      const int   f   = fc + fl;
      const float* w1r = w1 + (size_t)f * QQ;      // two consecutive 8-float rows
      float h0 = b1[f];
      float h1 = b1[f + 1];
#pragma unroll
      for (int q = 0; q < QQ; ++q) {
        const float zq = zT[m][q];
        h0 = fmaf(zq, w1r[q],      h0);
        h1 = fmaf(zq, w1r[QQ + q], h1);
      }
      h0 = fmaxf(h0, 0.0f);
      h1 = fmaxf(h1, 0.0f);
      v2h pk; pk[0] = (_Float16)h0; pk[1] = (_Float16)h1;
      *(v2h*)&hT[m][fl] = pk;
    }
    __syncthreads();

    // ---- Phase 2: WMMA  acc += A(hT) x B(w2h), 4x8 register blocking ----
#pragma unroll
    for (int ks = 0; ks < KCH; ks += 32) {
      // A fragments (ISA 16-bit A layout): lane<16 -> K {0..7,16..23}, lane>=16 -> K {8..15,24..31}
      const int kb = ks + (laneHi << 3);
      v16h a[4];
#pragma unroll
      for (int mi = 0; mi < 4; ++mi) {
        const int arow = mBase + mi * 16 + lane16;
        union { v16h v; v8h h[2]; } ua;
        ua.h[0] = *(const v8h*)&hT[arow][kb];
        ua.h[1] = *(const v8h*)&hT[arow][kb + 16];
        a[mi] = ua.v;
      }

#pragma unroll
      for (int j = 0; j < 8; ++j) {
        // B fragment (same K order as A): lane<16 -> K {0..7,16..23}, lane>=16 -> K {8..15,24..31}; col = lane16
        const _Float16* bp = w2h + (size_t)(eQ + j * 16 + lane16) * FFN
                                 + fc + ks + (laneHi << 3);
        union { v16h v; v8h h[2]; } ub;
        ub.h[0] = *(const v8h*)bp;
        ub.h[1] = *(const v8h*)(bp + 16);
        const v16h b = ub.v;
#pragma unroll
        for (int mi = 0; mi < 4; ++mi) {
          acc[mi][j] = __builtin_amdgcn_wmma_f32_16x16x32_f16(
              false, a[mi], false, b, (short)0, acc[mi][j], false, false);
        }
      }
    }
    __syncthreads();
  }

  // ---- Phase 3: epilogue, + b2, non-temporal f32 stores ----
  // C/D layout: VGPR r -> M = r + 8*laneHi, N = lane16
#pragma unroll
  for (int j = 0; j < 8; ++j) {
    const int col  = eQ + j * 16 + lane16;
    const float bv = b2[col];
#pragma unroll
    for (int mi = 0; mi < 4; ++mi) {
      const int orow = m0 + mBase + mi * 16 + laneHi * 8;
#pragma unroll
      for (int r = 0; r < 8; ++r) {
        __builtin_nontemporal_store(acc[mi][j][r] + bv,
                                    out + (size_t)(orow + r) * EE + col);
      }
    }
  }
}

// ---------------------------------------------------------------------------
extern "C" void kernel_launch(void* const* d_in, const int* in_sizes, int n_in,
                              void* d_out, int out_size, void* d_ws, size_t ws_size,
                              hipStream_t stream) {
  (void)in_sizes; (void)n_in; (void)out_size; (void)ws_size;
  const float* x  = (const float*)d_in[0];
  const float* ry = (const float*)d_in[1];
  const float* w1 = (const float*)d_in[2];
  const float* b1 = (const float*)d_in[3];
  const float* w2 = (const float*)d_in[4];
  const float* b2 = (const float*)d_in[5];
  float* out = (float*)d_out;
  _Float16* w2h = (_Float16*)d_ws;   // 512*2048*2 = 2 MB of workspace

  cvt_w2_kernel<<<(EE * FFN) / 1024, 256, 0, stream>>>(w2, w2h);
  ffnq_kernel<<<MTOT / MTILE, 256, 0, stream>>>(x, ry, w1, b1, w2h, b2, out);
}
